// InstanSeg_SAM_67276367725234
// MI455X (gfx1250) — matrix operand order repaired and checked by reading, not compiled, not run on hardware
//
#include <hip/hip_runtime.h>
#include <math.h>

// ---------------- CDNA5 WMMA types ----------------
typedef __attribute__((ext_vector_type(16))) __bf16 v16bf;
typedef __attribute__((ext_vector_type(8)))  float  v8f;

#define HEADS 16
#define RMAXR 63

#define FLAG_GELU   1
#define FLAG_TRANSB 2

// fp32 -> bf16 bits, round-to-nearest-even
__device__ __forceinline__ unsigned short f2bf(float f) {
  union { float f; unsigned u; } x; x.f = f;
  unsigned u = x.u;
  unsigned r = u + 0x7FFFu + ((u >> 16) & 1u);
  return (unsigned short)(r >> 16);
}

union FragU { uint4 q[2]; v16bf v; };

// LDS tile geometry: 128 rows x 32 K (bf16), row stride 80 bytes (pad => conflict-free frag loads)
#define LROW 80

// Stage a row-major [rows x K] fp32 source into LDS as bf16 [128 x 32] (A tiles, and B tiles when transB)
__device__ __forceinline__ void stage_rows(unsigned char* lds, const float* __restrict__ G, int ldg,
                                           int rowBase, int rMax, int kt, int K, int tid) {
#pragma unroll
  for (int it = 0; it < 4; ++it) {
    int i = tid + it * 256;           // 128 rows * 8 float4 groups
    int r = i >> 3, c4 = (i & 7) << 2;
    int grow = rowBase + r, gk = kt + c4;
    float f0 = 0.f, f1 = 0.f, f2 = 0.f, f3 = 0.f;
    if (grow < rMax) {
      const float* p = G + (size_t)grow * (size_t)ldg + gk;
      if (gk + 4 <= K) {
        float4 t4 = *(const float4*)p;
        f0 = t4.x; f1 = t4.y; f2 = t4.z; f3 = t4.w;
      } else {
        if (gk     < K) f0 = p[0];
        if (gk + 1 < K) f1 = p[1];
        if (gk + 2 < K) f2 = p[2];
        if (gk + 3 < K) f3 = p[3];
      }
    }
    uint2 u;
    u.x = (unsigned)f2bf(f0) | ((unsigned)f2bf(f1) << 16);
    u.y = (unsigned)f2bf(f2) | ((unsigned)f2bf(f3) << 16);
    *(uint2*)(lds + r * LROW + (c4 << 1)) = u;
  }
}

// Stage row-major B [K x N] fp32 into LDS transposed as [n][k] bf16 (so per-lane column reads are contiguous)
__device__ __forceinline__ void stage_bt(unsigned char* lds, const float* __restrict__ G, int ldg,
                                         int nBase, int nMax, int kt, int K, int tid) {
  unsigned short* l = (unsigned short*)lds;
#pragma unroll
  for (int it = 0; it < 4; ++it) {
    int i = tid + it * 256;           // 32 k * 32 n-groups
    int k = i >> 5, n4 = (i & 31) << 2;
    int gk = kt + k;
#pragma unroll
    for (int j = 0; j < 4; ++j) {
      int gn = nBase + n4 + j;
      float f = 0.f;
      if (gk < K && gn < nMax) f = G[(size_t)gk * (size_t)ldg + gn];
      l[(size_t)(n4 + j) * (LROW / 2) + k] = f2bf(f);
    }
  }
}

// A fragment (16x32 bf16): lanes 0-15 row M=lane, K 0..7 & 16..23 ; lanes 16-31 row M=lane-16, K 8..15 & 24..31
__device__ __forceinline__ v16bf loadFragA(const unsigned char* lds, int rowBase, int lane) {
  int l = lane & 15, hi = lane >> 4;
  const unsigned char* p = lds + (size_t)(rowBase + l) * LROW + hi * 16;
  FragU u;
  u.q[0] = *(const uint4*)p;
  u.q[1] = *(const uint4*)(p + 32);
  return u.v;
}

// B fragment (32x16 bf16), LDS holds [n][k]: lanes 0-15 col N=lane K 0..7 & 16..23; lanes 16-31 col N=lane-16 K 8..15 & 24..31
__device__ __forceinline__ v16bf loadFragB(const unsigned char* lds, int colBase, int lane) {
  int l = lane & 15, hi = lane >> 4;
  const unsigned char* p = lds + (size_t)(colBase + l) * LROW + hi * 16;
  FragU u;
  u.q[0] = *(const uint4*)p;
  u.q[1] = *(const uint4*)(p + 32);
  return u.v;
}

// Generic batched GEMM: C[z] = epilogue(alpha * A[z] @ B[z]) ; epilogue: +bias, +relpos, GELU, +residual
__global__ __launch_bounds__(256)
void gemm_bf16_k(const float* __restrict__ A, const float* __restrict__ B,
                 const float* __restrict__ bias, const float* __restrict__ R,
                 const float* __restrict__ relh, const float* __restrict__ relw,
                 float* __restrict__ C,
                 int M, int N, int K, int lda, int ldb, int ldc,
                 long long sA, long long sB, long long sC,
                 float alpha, int flags, int relW) {
  __shared__ __align__(16) unsigned char sAm[128 * LROW];
  __shared__ __align__(16) unsigned char sBm[128 * LROW];
  const int tid  = threadIdx.x;
  const int lane = tid & 31, wave = tid >> 5;
  const int wm = wave & 3, wn = wave >> 2;        // 4x2 wave grid -> 32x64 per wave
  const int m0 = blockIdx.y * 128, n0 = blockIdx.x * 128;
  const int z  = blockIdx.z;
  A += (long long)z * sA;
  B += (long long)z * sB;
  C += (long long)z * sC;

  const v8f zero8 = {0.f, 0.f, 0.f, 0.f, 0.f, 0.f, 0.f, 0.f};
  v8f acc[2][4];
#pragma unroll
  for (int a_ = 0; a_ < 2; ++a_)
#pragma unroll
    for (int b_ = 0; b_ < 4; ++b_) acc[a_][b_] = zero8;

  for (int kt = 0; kt < K; kt += 32) {
    if (kt + 32 < K)  // speculative prefetch of next A K-slab (global_prefetch_b8)
      __builtin_prefetch(A + (size_t)(m0 + (tid & 127)) * (size_t)lda + kt + 32, 0, 0);
    stage_rows(sAm, A, lda, m0, M, kt, K, tid);
    if (flags & FLAG_TRANSB) stage_rows(sBm, B, ldb, n0, N, kt, K, tid);
    else                     stage_bt(sBm, B, ldb, n0, N, kt, K, tid);
    __syncthreads();
    v16bf af0 = loadFragA(sAm, wm * 32,      lane);
    v16bf af1 = loadFragA(sAm, wm * 32 + 16, lane);
#pragma unroll
    for (int tn = 0; tn < 4; ++tn) {
      v16bf bfr = loadFragB(sBm, wn * 64 + tn * 16, lane);
      acc[0][tn] = __builtin_amdgcn_wmma_f32_16x16x32_bf16(false, af0, false, bfr, (short)0, acc[0][tn], false, false);
      acc[1][tn] = __builtin_amdgcn_wmma_f32_16x16x32_bf16(false, af1, false, bfr, (short)0, acc[1][tn], false, false);
    }
    __syncthreads();
  }

  // Epilogue. C/D layout: VGPR r -> M = r + (lane>=16 ? 8 : 0), N = lane&15 within 16x16 tile.
  const int l = lane & 15, hi = lane >> 4;
#pragma unroll
  for (int tm = 0; tm < 2; ++tm) {
#pragma unroll
    for (int tn = 0; tn < 4; ++tn) {
      int col = n0 + wn * 64 + tn * 16 + l;
      if (col >= N) continue;
#pragma unroll
      for (int r = 0; r < 8; ++r) {
        int row = m0 + wm * 32 + tm * 16 + hi * 8 + r;
        if (row >= M) continue;
        float v = alpha * acc[tm][tn][r];
        if (bias) v += bias[col];
        if (relh) {
          size_t base = ((size_t)z * (size_t)M + (size_t)row) * (size_t)relW;
          v += relh[base + col / relW] + relw[base + col % relW];
        }
        if (flags & FLAG_GELU) v = 0.5f * v * (1.f + erff(v * 0.70710678118654752f));
        size_t ci = (size_t)row * (size_t)ldc + col;
        if (R) v += R[ci];
        C[ci] = v;
      }
    }
  }
}

// ---------------- LayerNorm over last dim (also final NCHW-transposed write) ----------------
__global__ __launch_bounds__(256)
void ln_tok_k(const float* __restrict__ X, const float* __restrict__ g, const float* __restrict__ b,
              float* __restrict__ Y, int C, int transOut, int T) {
  int t = blockIdx.x;
  const float* x = X + (size_t)t * C;
  __shared__ float red[256];
  float s = 0.f;
  for (int c = threadIdx.x; c < C; c += 256) s += x[c];
  red[threadIdx.x] = s; __syncthreads();
  for (int off = 128; off; off >>= 1) { if (threadIdx.x < off) red[threadIdx.x] += red[threadIdx.x + off]; __syncthreads(); }
  float mu = red[0] / (float)C; __syncthreads();
  float v = 0.f;
  for (int c = threadIdx.x; c < C; c += 256) { float d = x[c] - mu; v += d * d; }
  red[threadIdx.x] = v; __syncthreads();
  for (int off = 128; off; off >>= 1) { if (threadIdx.x < off) red[threadIdx.x] += red[threadIdx.x + off]; __syncthreads(); }
  float rs = rsqrtf(red[0] / (float)C + 1e-6f);
  for (int c = threadIdx.x; c < C; c += 256) {
    float y = (x[c] - mu) * rs * g[c] + b[c];
    if (transOut) Y[(size_t)c * T + t] = y;
    else          Y[(size_t)t * C + c] = y;
  }
}

// ---------------- Row softmax ----------------
__global__ __launch_bounds__(256)
void softmax_k(float* __restrict__ S, int N) {
  float* row = S + (size_t)blockIdx.x * (size_t)N;
  __shared__ float red[256];
  float m = -3.4e38f;
  for (int i = threadIdx.x; i < N; i += 256) m = fmaxf(m, row[i]);
  red[threadIdx.x] = m; __syncthreads();
  for (int off = 128; off; off >>= 1) { if (threadIdx.x < off) red[threadIdx.x] = fmaxf(red[threadIdx.x], red[threadIdx.x + off]); __syncthreads(); }
  m = red[0]; __syncthreads();
  float s = 0.f;
  for (int i = threadIdx.x; i < N; i += 256) { float e = expf(row[i] - m); row[i] = e; s += e; }
  red[threadIdx.x] = s; __syncthreads();
  for (int off = 128; off; off >>= 1) { if (threadIdx.x < off) red[threadIdx.x] += red[threadIdx.x + off]; __syncthreads(); }
  float inv = 1.f / red[0];
  for (int i = threadIdx.x; i < N; i += 256) row[i] *= inv;
}

// ---------------- Decomposed rel-pos bias tables: tab[zi][q][k*] = dot(Q[q], R[coord - k + S - 1]) ----------------
__global__ __launch_bounds__(256)
void relbias_k(const float* __restrict__ qkvB, const float* __restrict__ Rt,
               float* __restrict__ tab, int N, int S, int h0, int useH, int total) {
  int idx = blockIdx.x * 256 + threadIdx.x;
  if (idx >= total) return;
  int kh = idx % S; int rest = idx / S; int q = rest % N; int zi = rest / N;
  int coord = useH ? (q / S) : (q % S);
  const float* Rr = Rt + (size_t)(coord - kh + S - 1) * 64;
  const float* Q  = qkvB + (size_t)q * 3072 + (size_t)(h0 + zi) * 64;
  float s = 0.f;
#pragma unroll
  for (int c = 0; c < 64; ++c) s += Q[c] * Rr[c];
  tab[idx] = s;
}

// ---------------- Data-movement kernels ----------------
__global__ __launch_bounds__(256)
void patch_im2col_k(const float* __restrict__ x, float* __restrict__ out) {
  int idx = blockIdx.x * 256 + threadIdx.x;
  if (idx >= 1024 * 768) return;
  int k = idx % 768, p = idx / 768;
  int py = p >> 5, px = p & 31;
  int hh = k / 48, rem = k % 48, ww = rem / 3, ci = rem % 3;
  out[idx] = x[(size_t)ci * 512 * 512 + (size_t)(py * 16 + hh) * 512 + (px * 16 + ww)];
}

__global__ __launch_bounds__(256)
void win_part_k(const float* __restrict__ a, float* __restrict__ w) {
  int idx = blockIdx.x * 256 + threadIdx.x;
  if (idx >= 9 * 196 * 1024) return;
  int c = idx & 1023, t = idx >> 10;
  int win = t / 196, l = t % 196;
  int wy = win / 3, wx = win % 3, ly = l / 14, lx = l % 14;
  int gy = wy * 14 + ly, gx = wx * 14 + lx;
  float v = 0.f;
  if (gy < 32 && gx < 32) v = a[(size_t)(gy * 32 + gx) * 1024 + c];
  w[idx] = v;
}

__global__ __launch_bounds__(256)
void win_unpart_k(const float* __restrict__ w, float* __restrict__ out) {
  int idx = blockIdx.x * 256 + threadIdx.x;
  if (idx >= 1024 * 1024) return;
  int c = idx & 1023, p = idx >> 10;
  int gy = p >> 5, gx = p & 31;
  int wy = gy / 14, ly = gy % 14, wx = gx / 14, lx = gx % 14;
  int t = (wy * 3 + wx) * 196 + ly * 14 + lx;
  out[idx] = w[(size_t)t * 1024 + c];
}

__global__ __launch_bounds__(256)
void neck_im2col_k(const float* __restrict__ in, float* __restrict__ out) {
  int idx = blockIdx.x * 256 + threadIdx.x;
  if (idx >= 1024 * 2304) return;
  int k = idx % 2304, p = idx / 2304;
  int r = p >> 5, c = p & 31;
  int ky = k / 768, rem = k % 768, kx = rem / 256, ci = rem & 255;
  int rr = r + ky - 1, cc = c + kx - 1;
  float v = 0.f;
  if (rr >= 0 && rr < 32 && cc >= 0 && cc < 32) v = in[(size_t)(rr * 32 + cc) * 256 + ci];
  out[idx] = v;
}

// ---------------- Host orchestration ----------------
extern "C" void kernel_launch(void* const* d_in, const int* in_sizes, int n_in,
                              void* d_out, int out_size, void* d_ws, size_t ws_size,
                              hipStream_t stream) {
  (void)in_sizes; (void)n_in; (void)out_size; (void)ws_size;

  const float* x        = (const float*)d_in[0];
  const float* patch_w  = (const float*)d_in[1];
  const float* patch_b  = (const float*)d_in[2];
  const float* ln1_g    = (const float*)d_in[3];
  const float* ln1_b    = (const float*)d_in[4];
  const float* qkv_w    = (const float*)d_in[5];
  const float* qkv_b    = (const float*)d_in[6];
  const float* proj_w   = (const float*)d_in[7];
  const float* proj_b   = (const float*)d_in[8];
  const float* rel_h    = (const float*)d_in[9];
  const float* rel_w    = (const float*)d_in[10];
  const float* ln2_g    = (const float*)d_in[11];
  const float* ln2_b    = (const float*)d_in[12];
  const float* mlp_w1   = (const float*)d_in[13];
  const float* mlp_b1   = (const float*)d_in[14];
  const float* mlp_w2   = (const float*)d_in[15];
  const float* mlp_b2   = (const float*)d_in[16];
  const float* neck_w1  = (const float*)d_in[17];
  const float* nln1_g   = (const float*)d_in[18];
  const float* nln1_b   = (const float*)d_in[19];
  const float* neck_w2  = (const float*)d_in[20];
  const float* nln2_g   = (const float*)d_in[21];
  const float* nln2_b   = (const float*)d_in[22];
  float* out = (float*)d_out;

  float* W = (float*)d_ws;
  size_t off = 0;
  auto alloc = [&](size_t n) { size_t r = off; off += (n + 255) & ~(size_t)255; return r; };
  const size_t o_h   = alloc((size_t)1024 * 1024);
  const size_t o_a   = alloc((size_t)1024 * 1024);
  const size_t o_w   = alloc((size_t)1764 * 1024);
  const size_t o_qkv = alloc((size_t)1764 * 3072);
  const size_t o_s   = alloc((size_t)4 * 1024 * 1024);  // score chunk (4 heads global / 16 heads window)
  const size_t o_rh  = alloc((size_t)131072);
  const size_t o_rw  = alloc((size_t)131072);
  const size_t o_o   = alloc((size_t)1764 * 1024);
  const size_t o_t   = alloc((size_t)1024 * 4096);      // mlp hidden / im2col / unpartition scratch
  const size_t o_n1  = alloc((size_t)1024 * 256);
  const size_t o_n2  = alloc((size_t)1024 * 256);

  auto gemm = [&](const float* A, const float* B, const float* bias, const float* Rres,
                  const float* rh, const float* rw, float* C,
                  int M, int N, int K, int lda, int ldb, int ldc,
                  long long sA, long long sB, long long sC, int Z,
                  float alpha, int flags, int relW) {
    dim3 g((unsigned)((N + 127) / 128), (unsigned)((M + 127) / 128), (unsigned)Z);
    gemm_bf16_k<<<g, dim3(256), 0, stream>>>(A, B, bias, Rres, rh, rw, C,
                                             M, N, K, lda, ldb, ldc, sA, sB, sC, alpha, flags, relW);
  };
  auto ew = [&](int total) { return dim3((unsigned)((total + 255) / 256)); };

  // ---- patch embed: im2col (1024x768) then GEMM with HWIO weights (768x1024) ----
  patch_im2col_k<<<ew(1024 * 768), 256, 0, stream>>>(x, W + o_t);
  gemm(W + o_t, patch_w, patch_b, nullptr, nullptr, nullptr, W + o_h,
       1024, 1024, 768, 768, 1024, 1024, 0, 0, 0, 1, 1.f, 0, 1);

  // ---- transformer blocks ----
  for (int i = 0; i < 4; ++i) {
    const bool glob = (i == 3);
    const int S  = glob ? 32 : 14;
    const int N  = S * S;                 // tokens per attention batch
    const int NB = glob ? 1 : 9;          // attention batches (windows)
    const int Mq = NB * N;
    const int HC = glob ? 4 : 16;         // heads per chunk

    ln_tok_k<<<1024, 256, 0, stream>>>(W + o_h, ln1_g + i * 1024, ln1_b + i * 1024, W + o_a, 1024, 0, 0);

    const float* attIn = W + o_a;
    if (!glob) {
      win_part_k<<<ew(9 * 196 * 1024), 256, 0, stream>>>(W + o_a, W + o_w);
      attIn = W + o_w;
    }

    // qkv = attIn @ qkv_w + qkv_b   (Mq x 3072)
    gemm(attIn, qkv_w + (size_t)i * 1024 * 3072, qkv_b + (size_t)i * 3072, nullptr, nullptr, nullptr,
         W + o_qkv, Mq, 3072, 1024, 1024, 3072, 3072, 0, 0, 0, 1, 1.f, 0, 1);

    const float* Rh = rel_h + (size_t)i * RMAXR * 64;
    const float* Rw = rel_w + (size_t)i * RMAXR * 64;

    for (int b = 0; b < NB; ++b) {
      const float* qkvB = W + o_qkv + (size_t)b * N * 3072;
      for (int h0 = 0; h0 < HEADS; h0 += HC) {
        const int total = HC * N * S;
        relbias_k<<<ew(total), 256, 0, stream>>>(qkvB, Rh, W + o_rh, N, S, h0, 1, total);
        relbias_k<<<ew(total), 256, 0, stream>>>(qkvB, Rw, W + o_rw, N, S, h0, 0, total);
        // S = 0.125 * Q @ K^T + relh + relw   (batched over HC heads)
        gemm(qkvB + h0 * 64, qkvB + 1024 + h0 * 64, nullptr, nullptr, W + o_rh, W + o_rw,
             W + o_s, N, N, 64, 3072, 3072, N,
             64, 64, (long long)N * N, HC, 0.125f, FLAG_TRANSB, S);
        softmax_k<<<(unsigned)(HC * N), 256, 0, stream>>>(W + o_s, N);
        // O = P @ V  -> token-major [token][head*64 + c]
        gemm(W + o_s, qkvB + 2048 + h0 * 64, nullptr, nullptr, nullptr, nullptr,
             W + o_o + (size_t)b * N * 1024 + h0 * 64, N, 64, N, N, 3072, 1024,
             (long long)N * N, 64, 64, HC, 1.f, 0, 1);
      }
    }

    const float* projIn = W + o_o;
    if (!glob) {
      win_unpart_k<<<ew(1024 * 1024), 256, 0, stream>>>(W + o_o, W + o_t);
      projIn = W + o_t;
    }
    // h = h + O @ proj_w + proj_b
    gemm(projIn, proj_w + (size_t)i * 1024 * 1024, proj_b + (size_t)i * 1024, W + o_h,
         nullptr, nullptr, W + o_h, 1024, 1024, 1024, 1024, 1024, 1024, 0, 0, 0, 1, 1.f, 0, 1);

    // MLP
    ln_tok_k<<<1024, 256, 0, stream>>>(W + o_h, ln2_g + i * 1024, ln2_b + i * 1024, W + o_a, 1024, 0, 0);
    gemm(W + o_a, mlp_w1 + (size_t)i * 1024 * 4096, mlp_b1 + (size_t)i * 4096, nullptr, nullptr, nullptr,
         W + o_t, 1024, 4096, 1024, 1024, 4096, 4096, 0, 0, 0, 1, 1.f, FLAG_GELU, 1);
    gemm(W + o_t, mlp_w2 + (size_t)i * 4096 * 1024, mlp_b2 + (size_t)i * 1024, W + o_h,
         nullptr, nullptr, W + o_h, 1024, 1024, 4096, 4096, 1024, 1024, 0, 0, 0, 1, 1.f, 0, 1);
  }

  // ---- neck ----
  // 1x1 conv: (1024x1024) @ (1024x256)
  gemm(W + o_h, neck_w1, nullptr, nullptr, nullptr, nullptr, W + o_n1,
       1024, 256, 1024, 1024, 256, 256, 0, 0, 0, 1, 1.f, 0, 1);
  ln_tok_k<<<1024, 256, 0, stream>>>(W + o_n1, nln1_g, nln1_b, W + o_n2, 256, 0, 0);
  // 3x3 SAME conv via im2col: (1024x2304) @ (2304x256)
  neck_im2col_k<<<ew(1024 * 2304), 256, 0, stream>>>(W + o_n2, W + o_t);
  gemm(W + o_t, neck_w2, nullptr, nullptr, nullptr, nullptr, W + o_n1,
       1024, 256, 2304, 2304, 256, 256, 0, 0, 0, 1, 1.f, 0, 1);
  // final LN2d, written transposed to NCHW (256,32,32)
  ln_tok_k<<<1024, 256, 0, stream>>>(W + o_n1, nln2_g, nln2_b, out, 256, 1, 1024);
}
